// StableGroupedVarPatchEmbed_18176301597286
// MI455X (gfx1250) — matrix operand order repaired and checked by reading, not compiled, not run on hardware
//
#include <hip/hip_runtime.h>

typedef __attribute__((ext_vector_type(16))) _Float16 v16h;
typedef __attribute__((ext_vector_type(8)))  _Float16 v8h;
typedef __attribute__((ext_vector_type(8)))  float    v8f;
typedef __attribute__((ext_vector_type(4)))  float    v4f;

// Problem constants (fixed by reference)
#define CB   32      // batch
#define CV   8       // variables used
#define CH   256     // H
#define CW   256     // W
#define CD   1024    // embed dim
#define CK   256     // K = 16*16 patch
#define CP   16      // patch side
#define CPATCHES 256 // patches per (b,v) = 16*16

// Tiling
#define BM 128
#define BN 128
#define BK 32
#define LDK 40      // padded LDS pitch in halves (80B -> conflict-spread)

__global__ __launch_bounds__(256)
void patch_embed_wmma(const float* __restrict__ x,
                      const float* __restrict__ Wc,
                      const float* __restrict__ bias,
                      const int*   __restrict__ vars,
                      float* __restrict__ out)
{
    __shared__ __align__(16) _Float16 As[BM * LDK];
    __shared__ __align__(16) _Float16 Bs[BN * LDK];

    const int v      = blockIdx.z;
    const int tileM  = blockIdx.y * BM;          // M total = CB*CPATCHES = 8192
    const int tileN  = blockIdx.x * BN;          // N total = CD = 1024
    const int var_id = vars[v];

    const int t      = threadIdx.x;
    const int lane   = t & 31;
    const int wid    = t >> 5;
    const int wave_m = wid >> 2;                 // 0..1 -> 64 rows
    const int wave_n = wid & 3;                  // 0..3 -> 32 cols

    // ---- global-load mapping: each thread fetches 16 contiguous floats ----
    const int lm  = t >> 1;                      // 0..127: row within tile
    const int lkh = t & 1;                       // which 16-elem half of BK

    // A row -> (b_idx, hh, ww)
    const int m_g   = tileM + lm;
    const int b_idx = m_g >> 8;
    const int l_in  = m_g & 255;
    const int hh    = l_in >> 4;
    const int ww    = l_in & 15;
    const float* xrow_base =
        x + ((size_t)(b_idx * CV + v) * CH + (size_t)hh * CP) * CW + (size_t)ww * CP;
    // per k-tile: row p = kt*2 + lkh  ->  + p*CW

    const float* wrow_base =
        Wc + ((size_t)var_id * CD + (size_t)(tileN + lm)) * CK + (size_t)lkh * 16;
    // per k-tile: + kt*BK

    // accumulators: 4 (M) x 2 (N) tiles of 16x16
    v8f acc[4][2];
#pragma unroll
    for (int i = 0; i < 4; ++i)
#pragma unroll
        for (int j = 0; j < 2; ++j)
            acc[i][j] = v8f{};

    const int mrow = lane & 15;
    const int ka   = (lane >> 4) * 8;    // A k-offset for upper/lower half-wave
    const int kb   = (lane >> 4) * 8;    // B k-offset

    union Frag { v16h v; v8h h[2]; };

    for (int kt = 0; kt < CK / BK; ++kt) {
        // ---- global -> regs (f32) ----
        const float* ga = xrow_base + (size_t)(kt * 2 + lkh) * CW;
        const float* gb = wrow_base + (size_t)kt * BK;
        v4f a0 = *(const v4f*)(ga + 0);
        v4f a1 = *(const v4f*)(ga + 4);
        v4f a2 = *(const v4f*)(ga + 8);
        v4f a3 = *(const v4f*)(ga + 12);
        v4f b0 = *(const v4f*)(gb + 0);
        v4f b1 = *(const v4f*)(gb + 4);
        v4f b2 = *(const v4f*)(gb + 8);
        v4f b3 = *(const v4f*)(gb + 12);

        __syncthreads();   // WAR: previous iteration's LDS reads done

        // ---- convert f32 -> f16, store to LDS (2x ds_store_b128 each) ----
        v8h pa0, pa1, pb0, pb1;
#pragma unroll
        for (int i = 0; i < 4; ++i) {
            pa0[i]     = (_Float16)a0[i];
            pa0[4 + i] = (_Float16)a1[i];
            pa1[i]     = (_Float16)a2[i];
            pa1[4 + i] = (_Float16)a3[i];
            pb0[i]     = (_Float16)b0[i];
            pb0[4 + i] = (_Float16)b1[i];
            pb1[i]     = (_Float16)b2[i];
            pb1[4 + i] = (_Float16)b3[i];
        }
        _Float16* as = &As[lm * LDK + lkh * 16];
        _Float16* bs = &Bs[lm * LDK + lkh * 16];
        *(v8h*)(as + 0) = pa0;
        *(v8h*)(as + 8) = pa1;
        *(v8h*)(bs + 0) = pb0;
        *(v8h*)(bs + 8) = pb1;

        __syncthreads();   // RAW: tile visible to all waves

        // ---- build fragments from LDS (exact CDNA5 wave32 layouts) ----
        const _Float16* abase = &As[(wave_m * 64) * LDK];
        const _Float16* bbase = &Bs[(wave_n * 32) * LDK];

        Frag afrag[4], bfrag[2];
#pragma unroll
        for (int i = 0; i < 4; ++i) {
            const _Float16* ap = abase + (i * 16 + mrow) * LDK + ka;
            afrag[i].h[0] = *(const v8h*)(ap);        // K = ka .. ka+7
            afrag[i].h[1] = *(const v8h*)(ap + 16);   // K = ka+16 .. ka+23
        }
#pragma unroll
        for (int j = 0; j < 2; ++j) {
            const _Float16* bp = bbase + (j * 16 + mrow) * LDK + kb;
            bfrag[j].h[0] = *(const v8h*)(bp);        // K = kb .. kb+7
            bfrag[j].h[1] = *(const v8h*)(bp + 16);   // K = kb+16 .. kb+23
        }

        // ---- 8x v_wmma_f32_16x16x32_f16 ----
#pragma unroll
        for (int i = 0; i < 4; ++i)
#pragma unroll
            for (int j = 0; j < 2; ++j)
                acc[i][j] = __builtin_amdgcn_wmma_f32_16x16x32_f16(
                    false, afrag[i].v, false, bfrag[j].v,
                    (short)0, acc[i][j], false, false);
    }

    // ---- epilogue: bias + store ----
    float bv[2];
#pragma unroll
    for (int j = 0; j < 2; ++j) {
        const int d = tileN + wave_n * 32 + j * 16 + (lane & 15);
        bv[j] = bias[(size_t)var_id * CD + d];
    }

#pragma unroll
    for (int i = 0; i < 4; ++i) {
        const int m0 = tileM + wave_m * 64 + i * 16 + (lane >> 4) * 8; // first row for this lane
        const int bb = m0 >> 8;          // batch index (constant across the 8 rows)
        const int l0 = m0 & 255;         // patch index within (b,v)
#pragma unroll
        for (int j = 0; j < 2; ++j) {
            const int d = tileN + wave_n * 32 + j * 16 + (lane & 15);
            float* op = out + (((size_t)(bb * CV + v) * CPATCHES + l0) * CD + d);
#pragma unroll
            for (int r = 0; r < 8; ++r)
                op[(size_t)r * CD] = acc[i][j][r] + bv[j];
        }
    }
}

extern "C" void kernel_launch(void* const* d_in, const int* in_sizes, int n_in,
                              void* d_out, int out_size, void* d_ws, size_t ws_size,
                              hipStream_t stream) {
    const float* x    = (const float*)d_in[0];
    const float* Wc   = (const float*)d_in[1];
    const float* bias = (const float*)d_in[2];
    const int*   vars = (const int*)d_in[3];
    float* out = (float*)d_out;

    dim3 grid(CD / BN,                    // 8  N-tiles
              (CB * CPATCHES) / BM,       // 64 M-tiles
              CV);                        // 8  variables
    dim3 block(256);
    patch_embed_wmma<<<grid, block, 0, stream>>>(x, Wc, bias, vars, out);
}
